// GraphConvolution_89154931130498
// MI455X (gfx1250) — matrix operand order repaired and checked by reading, not compiled, not run on hardware
//
#include <hip/hip_runtime.h>
#include <hip/hip_bf16.h>

// Problem constants (match reference)
#define NUMG   3
#define VJ     20
#define CIN    64
#define COUT   128
#define NBATCH 64
#define TLEN   300
#define KDIM   (NUMG*CIN)      // 192
#define TBLK   16
#define ROWS   (TBLK*VJ)       // 320 rows (t_local,w) per tile
#define ZSTR   200             // padded K stride (ushorts) to dodge bank conflicts
#define NTBLK  19              // ceil(300/16)
#define PLANE  (TLEN*VJ)       // 6000
#define NWG    (NBATCH*NTBLK)  // 1216 workgroups in kernel 1
#define PART_OFF ((size_t)NWG*256)
#define CNT_INV  (1.0f/384000.0f)   // 1/(N*T*V)

typedef __attribute__((ext_vector_type(16))) __bf16 v16bf;
typedef __attribute__((ext_vector_type(8)))  float  v8f;
typedef __attribute__((ext_vector_type(8)))  short  short8;

union ABPack {
    short8 h[2];
    unsigned short u16[16];
    v16bf v;
};

__device__ inline unsigned short f2bf(float f) {
    unsigned u = __builtin_bit_cast(unsigned, f);
    u += 0x7FFFu + ((u >> 16) & 1u);   // round to nearest even
    return (unsigned short)(u >> 16);
}

// ---------------------------------------------------------------------------
// Kernel 1: fused (adj*mask) graph contraction + per-graph 1x1 conv as WMMA
// GEMM, bias add, raw-y store, per-channel partial sums for BatchNorm.
// One workgroup = one (n, t-block of 16). 8 waves, each owns 16 channels.
// ---------------------------------------------------------------------------
extern "C" __global__ void __launch_bounds__(256)
gc_fused_wmma(const float* __restrict__ adj, const float* __restrict__ x,
              const float* __restrict__ mask, const float* __restrict__ W,
              const float* __restrict__ b, float* __restrict__ y,
              float* __restrict__ partials)
{
    extern __shared__ char smem[];
    unsigned short* zaS = (unsigned short*)smem;                       // [320][200] bf16
    float* aS  = (float*)(smem + (size_t)ROWS*ZSTR*2);                 // [3][20][20]
    float* stS = (float*)(smem + (size_t)ROWS*ZSTR*2 + NUMG*VJ*VJ*4);  // [256] sums|sumsq

    const int tid  = threadIdx.x;
    const int bid  = blockIdx.x;
    const int n    = bid / NTBLK;
    const int tblk = bid % NTBLK;

    // masked adjacency into LDS; zero per-channel stat accumulators
    for (int i = tid; i < NUMG*VJ*VJ; i += 256) aS[i] = adj[i] * mask[i];
    stS[tid] = 0.0f;
    __syncthreads();

    // ---- stage 1: za[m][k] = sum_v x[n,c,t,v]*a[g,v,w]; m=t_local*20+w, k=g*64+c
    // Each thread owns 2 units of (t_local, channel-pair): computes both c and
    // c+1 so the two bf16 results pack into one ds_store_b32. Prefetch the
    // next unit's rows while this unit's 2400 FMAs run (global_prefetch_b8).
    {
        const float* xr0[2];
        const float* xr1[2];
        for (int i = 0; i < 2; ++i) {
            int u  = tid + 256*i;          // 512 units
            int tl = u & 15;
            int cp = u >> 4;               // channel pair 0..31
            int t  = tblk*TBLK + tl;
            const float* base = x + (((size_t)n*CIN + 2*cp)*TLEN + t)*VJ;
            xr0[i] = (t < TLEN) ? base : nullptr;
            xr1[i] = (t < TLEN) ? base + (size_t)TLEN*VJ : nullptr;
        }
        if (xr0[0]) { __builtin_prefetch(xr0[0], 0, 1); __builtin_prefetch(xr1[0], 0, 1); }
        for (int i = 0; i < 2; ++i) {
            if (i == 0 && xr0[1]) { __builtin_prefetch(xr0[1], 0, 1); __builtin_prefetch(xr1[1], 0, 1); }
            int u  = tid + 256*i;
            int tl = u & 15;
            int cp = u >> 4;
            float xv0[VJ], xv1[VJ];
            if (xr0[i]) {
                const float4* p0 = (const float4*)xr0[i];
                const float4* p1 = (const float4*)xr1[i];
                #pragma unroll
                for (int q = 0; q < 5; ++q) {
                    float4 f0 = p0[q], f1 = p1[q];
                    xv0[4*q+0]=f0.x; xv0[4*q+1]=f0.y; xv0[4*q+2]=f0.z; xv0[4*q+3]=f0.w;
                    xv1[4*q+0]=f1.x; xv1[4*q+1]=f1.y; xv1[4*q+2]=f1.z; xv1[4*q+3]=f1.w;
                }
            } else {
                #pragma unroll
                for (int q = 0; q < VJ; ++q) { xv0[q] = 0.0f; xv1[q] = 0.0f; }
            }
            for (int g = 0; g < NUMG; ++g) {
                for (int w = 0; w < VJ; ++w) {
                    float s0 = 0.0f, s1 = 0.0f;
                    #pragma unroll
                    for (int v = 0; v < VJ; ++v) {
                        float av = aS[(g*VJ + v)*VJ + w];
                        s0 = fmaf(xv0[v], av, s0);
                        s1 = fmaf(xv1[v], av, s1);
                    }
                    unsigned pk = (unsigned)f2bf(s0) | ((unsigned)f2bf(s1) << 16);
                    *(unsigned*)(zaS + (tl*VJ + w)*ZSTR + g*CIN + 2*cp) = pk;  // ds_store_b32
                }
            }
        }
    }
    __syncthreads();

    // ---- stage 2: y[m][o] = sum_k za[m][k] * W2[k][o], K=192 = 6 x (16x16x32 bf16 WMMA)
    const int lane = tid & 31;
    const int wave = tid >> 5;                 // o-tile 0..7
    const int ocol = wave*16 + (lane & 15);
    const int kb   = (lane < 16) ? 0 : 8;      // B frag: lanes 0-15 K=0..7,16..23, 16-31 K=8..15,24..31

    v16bf Bfrag[6];
    #pragma unroll
    for (int kk = 0; kk < 6; ++kk) {
        ABPack u;
        #pragma unroll
        for (int e = 0; e < 16; ++e) {
            int k = kk*32 + kb + (e < 8 ? e : e + 8);
            int g = k >> 6, c = k & 63;
            u.u16[e] = f2bf(W[((size_t)g*COUT + ocol)*CIN + c]);
        }
        Bfrag[kk] = u.v;
        // Opaque barrier: force the packed bf16 fragment to be materialized
        // once here, so the f32->bf16 conversions cannot be sunk/remat'ed
        // into the WMMA loop (and the source W floats die).
        asm volatile("" : "+v"(Bfrag[kk]));
    }

    const float bsum  = b[ocol] + b[COUT + ocol] + b[2*COUT + ocol];
    const int   vrows = (tblk == NTBLK-1) ? (TLEN - (NTBLK-1)*TBLK)*VJ : ROWS; // 240 or 320
    float* yplane = y + ((size_t)n*COUT + ocol)*PLANE + (size_t)tblk*ROWS;

    const int ab = (lane < 16) ? 0 : 8;        // A frag K sub-offset per lane half
    float lsum = 0.0f, lsq = 0.0f;

    for (int rt = 0; rt < 20; ++rt) {
        const int arow = (rt*16 + (lane & 15)) * ZSTR;

        // Issue all 12 ds_load_b128 for the six A-fragments up front so the
        // six WMMAs can run back-to-back instead of waiting per K-step.
        ABPack ua[6];
        #pragma unroll
        for (int kk = 0; kk < 6; ++kk) {
            const short8* ap = (const short8*)(zaS + arow + kk*32 + ab);
            ua[kk].h[0] = ap[0];       // K = k0+ab    .. +7
            ua[kk].h[1] = ap[2];       // K = k0+ab+16 .. +23
        }

        v8f acc = {0.f,0.f,0.f,0.f,0.f,0.f,0.f,0.f};
        #pragma unroll
        for (int kk = 0; kk < 6; ++kk) {
            acc = __builtin_amdgcn_wmma_f32_16x16x32_bf16(
                      false, ua[kk].v, false, Bfrag[kk], (short)0, acc, false, false);
        }

        const int mbase = rt*16 + ab;          // D: lanes 0-15 -> M=e, lanes 16-31 -> M=8+e
        if (mbase + 3 < vrows) {
            float4 o0 = { acc[0]+bsum, acc[1]+bsum, acc[2]+bsum, acc[3]+bsum };
            *(float4*)(yplane + mbase) = o0;
            lsum += o0.x + o0.y + o0.z + o0.w;
            lsq  += o0.x*o0.x + o0.y*o0.y + o0.z*o0.z + o0.w*o0.w;
        }
        if (mbase + 7 < vrows) {
            float4 o1 = { acc[4]+bsum, acc[5]+bsum, acc[6]+bsum, acc[7]+bsum };
            *(float4*)(yplane + mbase + 4) = o1;
            lsum += o1.x + o1.y + o1.z + o1.w;
            lsq  += o1.x*o1.x + o1.y*o1.y + o1.z*o1.z + o1.w*o1.w;
        }
    }

    atomicAdd(&stS[ocol], lsum);          // ds_add_f32
    atomicAdd(&stS[COUT + ocol], lsq);
    __syncthreads();
    partials[(size_t)bid*256 + tid] = stS[tid];   // written fresh each call
}

// ---------------------------------------------------------------------------
// Kernel 2: per-channel reduction of 1216 workgroup partials -> scale/shift
// ---------------------------------------------------------------------------
extern "C" __global__ void __launch_bounds__(256)
gc_stats(const float* __restrict__ partials, const float* __restrict__ gamma,
         const float* __restrict__ beta, float* __restrict__ params)
{
    __shared__ float s0[256], s1[256];
    const int o = blockIdx.x;      // 0..127
    const int tid = threadIdx.x;
    float a = 0.0f, c = 0.0f;
    for (int wg = tid; wg < NWG; wg += 256) {
        a += partials[(size_t)wg*256 + o];
        c += partials[(size_t)wg*256 + COUT + o];
    }
    s0[tid] = a; s1[tid] = c;
    __syncthreads();
    for (int s = 128; s > 0; s >>= 1) {
        if (tid < s) { s0[tid] += s0[tid+s]; s1[tid] += s1[tid+s]; }
        __syncthreads();
    }
    if (tid == 0) {
        float mean = s0[0]*CNT_INV;
        float var  = s1[0]*CNT_INV - mean*mean;
        float rstd = rsqrtf(var + 1e-5f);
        float sc   = gamma[o]*rstd;
        params[o]        = sc;
        params[COUT + o] = beta[o] - mean*sc;
    }
}

// ---------------------------------------------------------------------------
// Kernel 3: y = relu(y*scale[o] + shift[o]), float4 vectorized, in place
// ---------------------------------------------------------------------------
extern "C" __global__ void __launch_bounds__(256)
gc_bn_relu(float* __restrict__ y, const float* __restrict__ params)
{
    size_t i = ((size_t)blockIdx.x*256 + threadIdx.x) * 4;
    int o = (int)((i / PLANE) & (COUT-1));    // plane size 6000 divisible by 4
    float sc = params[o], sh = params[COUT + o];
    float4 v = *(float4*)(y + i);
    v.x = fmaxf(fmaf(v.x, sc, sh), 0.0f);
    v.y = fmaxf(fmaf(v.y, sc, sh), 0.0f);
    v.z = fmaxf(fmaf(v.z, sc, sh), 0.0f);
    v.w = fmaxf(fmaf(v.w, sc, sh), 0.0f);
    *(float4*)(y + i) = v;
}

// ---------------------------------------------------------------------------
extern "C" void kernel_launch(void* const* d_in, const int* in_sizes, int n_in,
                              void* d_out, int out_size, void* d_ws, size_t ws_size,
                              hipStream_t stream)
{
    (void)in_sizes; (void)n_in; (void)out_size; (void)ws_size;
    const float* adj   = (const float*)d_in[0];
    const float* x     = (const float*)d_in[1];
    const float* mask  = (const float*)d_in[2];
    const float* W     = (const float*)d_in[3];
    const float* b     = (const float*)d_in[4];
    const float* gamma = (const float*)d_in[5];
    const float* beta  = (const float*)d_in[6];
    float* y        = (float*)d_out;
    float* partials = (float*)d_ws;              // NWG*256 floats
    float* params   = partials + PART_OFF;       // 256 floats: scale|shift

    const size_t lds1 = (size_t)ROWS*ZSTR*2 + NUMG*VJ*VJ*4 + 256*4;  // 133824 B

    gc_fused_wmma<<<NWG, 256, lds1, stream>>>(adj, x, mask, W, b, y, partials);
    gc_stats<<<COUT, 256, 0, stream>>>(partials, gamma, beta, params);
    // 64*128*300*20 / (256*4) = 48000 blocks, exact
    gc_bn_relu<<<48000, 256, 0, stream>>>(y, params);
}
